// Prior_43387759624361
// MI455X (gfx1250) — matrix operand order repaired and checked by reading, not compiled, not run on hardware
//
#include <hip/hip_runtime.h>

// Problem constants (match reference)
#define BB 512
#define TT 256
#define HH 512
#define H3 1536
#define NCC 512
#define SKEL 256
#define GEN 16

typedef __attribute__((ext_vector_type(8)))  __bf16 v8bf;
typedef __attribute__((ext_vector_type(16))) __bf16 v16bf;
typedef __attribute__((ext_vector_type(8)))  float  v8f;

__device__ __forceinline__ unsigned short f2bf(float f) {
    unsigned int u = __float_as_uint(f);
    unsigned int r = (u + 0x7FFFu + ((u >> 16) & 1u)) >> 16;
    return (unsigned short)r;
}

// ---------------------------------------------------------------------------
// fp32 -> bf16 raw-bits conversion (one-time, weights/embedding)
// ---------------------------------------------------------------------------
__global__ void cvt_bf16_kernel(const float* __restrict__ src,
                                unsigned short* __restrict__ dst, int n) {
    int i = blockIdx.x * blockDim.x + threadIdx.x;
    if (i < n) dst[i] = f2bf(src[i]);
}

__global__ void tok_init_kernel(int* tok) {
    int i = blockIdx.x * blockDim.x + threadIdx.x;
    if (i < BB) tok[i] = 0;
}

// h1_init = concat(note, genre) @ W_hid^T + b_hid   (one-time, small)
__global__ void hidden_init_kernel(const float* __restrict__ note,
                                   const float* __restrict__ genre,
                                   const float* __restrict__ Whid,
                                   const float* __restrict__ bhid,
                                   float* __restrict__ h1,
                                   unsigned short* __restrict__ h1bf) {
    int idx = blockIdx.x * blockDim.x + threadIdx.x;   // [0, B*H)
    int b = idx >> 9;          // / HH
    int j = idx & (HH - 1);
    const float* wr = Whid + (size_t)j * (SKEL + GEN);
    float acc = bhid[j];
    const float* nb = note + (size_t)b * SKEL;
    for (int k = 0; k < SKEL; ++k) acc += nb[k] * wr[k];
    const float* gb = genre + (size_t)b * GEN;
    for (int k = 0; k < GEN; ++k) acc += gb[k] * wr[SKEL + k];
    h1[idx] = acc;
    h1bf[idx] = f2bf(acc);
}

// ---------------------------------------------------------------------------
// Core WMMA GEMM body: C[16x64 strip](+bias) = A[M,K](bf16) * W[N,K]^T(bf16).
// One wave computes one 16x64 strip. Ping-pong double buffering (K-step 64,
// two alternating register sets) hides L2 load latency behind the WMMAs with
// no cross-iteration register copies. Requires K % 64 == 0 (K = 512 here).
// ---------------------------------------------------------------------------
__device__ __forceinline__
void wmma_gemm_body(const unsigned short* __restrict__ Abf,
                    const int* __restrict__ gather,
                    const unsigned short* __restrict__ Wbf,
                    const float* __restrict__ bias,
                    float* __restrict__ C,
                    int K, int ldc, int mt, int nb) {
    const int lane = threadIdx.x;
    const int half = lane >> 4;      // 0: lanes 0-15, 1: lanes 16-31
    const int lid  = lane & 15;

    // A fragment source row (A layout: both lane halves hold row M=lid)
    const int mrow = mt * 16 + lid;
    const int arow_idx = gather ? gather[mrow] : mrow;
    const unsigned short* arow = Abf + (size_t)arow_idx * K;

    // B fragment source rows: lane holds column N=lid, half selects K-range
    const unsigned short* brow[4];
#pragma unroll
    for (int j = 0; j < 4; ++j) {
        int n = (nb * 4 + j) * 16 + lid;
        brow[j] = Wbf + (size_t)n * K + half * 16;
    }

    v8f acc[4];
#pragma unroll
    for (int j = 0; j < 4; ++j) {
        int n = (nb * 4 + j) * 16 + lid;
        float bv = bias ? bias[n] : 0.0f;
#pragma unroll
        for (int r = 0; r < 8; ++r) acc[j][r] = bv;
    }

    // A 16x32 bf16 fragment: two 16-byte runs per lane
    auto loadA = [&](int k) -> v16bf {
        v8bf x0 = *(const v8bf*)(const void*)(arow + k + half * 16);
        v8bf x1 = *(const v8bf*)(const void*)(arow + k + half * 16 + 8);
        return __builtin_shufflevector(x0, x1, 0, 1, 2, 3, 4, 5, 6, 7,
                                       8, 9, 10, 11, 12, 13, 14, 15);
    };

    // ---- prologue: set A <- chunk 0 ----
    v16bf aA = loadA(0);
    v16bf bA[4], bB[4];
#pragma unroll
    for (int j = 0; j < 4; ++j)
        bA[j] = *(const v16bf*)(const void*)(brow[j]);

    for (int k0 = 0; k0 < K; k0 += 64) {
        // set B <- chunk k0+32 (loads overlap set-A WMMAs)
        v16bf aB = loadA(k0 + 32);
#pragma unroll
        for (int j = 0; j < 4; ++j)
            bB[j] = *(const v16bf*)(const void*)(brow[j] + k0 + 32);

#pragma unroll
        for (int j = 0; j < 4; ++j)
            acc[j] = __builtin_amdgcn_wmma_f32_16x16x32_bf16(
                false, aA, false, bA[j], (short)0, acc[j], false, false);

        if (k0 + 64 < K) {
            // set A <- chunk k0+64 (loads overlap set-B WMMAs)
            aA = loadA(k0 + 64);
#pragma unroll
            for (int j = 0; j < 4; ++j)
                bA[j] = *(const v16bf*)(const void*)(brow[j] + k0 + 64);
            __builtin_prefetch(brow[0] + k0 + 96, 0, 3);  // global_prefetch_b8
        }

#pragma unroll
        for (int j = 0; j < 4; ++j)
            acc[j] = __builtin_amdgcn_wmma_f32_16x16x32_bf16(
                false, aB, false, bB[j], (short)0, acc[j], false, false);
    }

    // C layout: lane<16 -> M=r, lane>=16 -> M=8+r; N = lid
#pragma unroll
    for (int j = 0; j < 4; ++j) {
        int nout = (nb * 4 + j) * 16 + lid;
#pragma unroll
        for (int r = 0; r < 8; ++r) {
            int m = mt * 16 + half * 8 + r;
            C[(size_t)m * ldc + nout] = acc[j][r];
        }
    }
}

// Single GEMM (used for the logits projection). grid = (N/64, M/16).
__global__ __launch_bounds__(32)
void wmma_gemm_kernel(const unsigned short* __restrict__ Abf,
                      const int* __restrict__ gather,
                      const unsigned short* __restrict__ Wbf,
                      const float* __restrict__ bias,
                      float* __restrict__ C, int K, int ldc) {
    wmma_gemm_body(Abf, gather, Wbf, bias, C, K, ldc, blockIdx.y, blockIdx.x);
}

// Fused pair of independent GEMMs (gi and gh of one GRU layer) selected by
// blockIdx.z — halves the launch count on the serial autoregressive path.
// grid = (N/64, M/16, 2).
__global__ __launch_bounds__(32)
void wmma_gemm_dual_kernel(const unsigned short* __restrict__ A0,
                           const int* __restrict__ gather0,
                           const unsigned short* __restrict__ W0,
                           const float* __restrict__ bias0,
                           float* __restrict__ C0,
                           const unsigned short* __restrict__ A1,
                           const unsigned short* __restrict__ W1,
                           const float* __restrict__ bias1,
                           float* __restrict__ C1,
                           int K, int ldc) {
    if (blockIdx.z == 0)
        wmma_gemm_body(A0, gather0, W0, bias0, C0, K, ldc, blockIdx.y, blockIdx.x);
    else
        wmma_gemm_body(A1, nullptr, W1, bias1, C1, K, ldc, blockIdx.y, blockIdx.x);
}

// ---------------------------------------------------------------------------
// GRU gate math: h = (1-z)*n + z*h_prev, torch GRUCell gate order [r,z,n].
// Writes fp32 h (for next gate math) and bf16 h (for next WMMA GEMM).
// h2dup non-null only at t==0 (hx[1] = hx[0]).
// ---------------------------------------------------------------------------
__global__ void gru_gates_kernel(const float* __restrict__ gi,
                                 const float* __restrict__ gh,
                                 float* h,               // in: h_prev, out: h_new
                                 unsigned short* hbf,
                                 float* h2dup, unsigned short* h2dupbf) {
    int idx = blockIdx.x * blockDim.x + threadIdx.x;   // [0, B*H)
    int b = idx >> 9;
    int j = idx & (HH - 1);
    size_t g = (size_t)b * H3 + j;
    float r = 1.0f / (1.0f + __expf(-(gi[g] + gh[g])));
    float z = 1.0f / (1.0f + __expf(-(gi[g + HH] + gh[g + HH])));
    float n = tanhf(gi[g + 2 * HH] + r * gh[g + 2 * HH]);
    float hn = (1.0f - z) * n + z * h[idx];
    h[idx] = hn;
    hbf[idx] = f2bf(hn);
    if (h2dup) { h2dup[idx] = hn; h2dupbf[idx] = f2bf(hn); }
}

// ---------------------------------------------------------------------------
// Categorical sampling via Gumbel-max (own counter-based hash RNG).
// One 256-thread block per batch row over NC=512 logits.
// ---------------------------------------------------------------------------
__device__ __forceinline__ unsigned int hash32(unsigned int x) {
    x ^= x >> 16; x *= 0x7feb352du;
    x ^= x >> 15; x *= 0x846ca68bu;
    x ^= x >> 16;
    return x;
}

__global__ void sample_kernel(const float* __restrict__ out_base, int t,
                              int* __restrict__ tok, float* __restrict__ pred) {
    int b = blockIdx.x;
    const float* row = out_base + ((size_t)b * TT + t) * NCC;
    __shared__ float sval[256];
    __shared__ int   sidx[256];
    float best = -3.4e38f; int bi = 0;
    for (int c = threadIdx.x; c < NCC; c += 256) {
        unsigned int hsh = hash32(((unsigned)t * 0x9E3779B9u) ^
                                  ((unsigned)b * 0x85EBCA6Bu) ^ (unsigned)c);
        float u = ((float)hsh + 1.0f) * 2.3283064e-10f;   // (0,1]
        float gum = -logf(-logf(u));
        float v = row[c] + gum;
        if (v > best) { best = v; bi = c; }
    }
    sval[threadIdx.x] = best; sidx[threadIdx.x] = bi;
    __syncthreads();
    for (int s = 128; s > 0; s >>= 1) {
        if (threadIdx.x < s && sval[threadIdx.x + s] > sval[threadIdx.x]) {
            sval[threadIdx.x] = sval[threadIdx.x + s];
            sidx[threadIdx.x] = sidx[threadIdx.x + s];
        }
        __syncthreads();
    }
    if (threadIdx.x == 0) {
        tok[b] = sidx[0];
        pred[(size_t)b * TT + t] = (float)sidx[0];
    }
}

// ---------------------------------------------------------------------------
extern "C" void kernel_launch(void* const* d_in, const int* in_sizes, int n_in,
                              void* d_out, int out_size, void* d_ws, size_t ws_size,
                              hipStream_t stream) {
    // inputs (setup_inputs order)
    const float* note   = (const float*)d_in[1];
    const float* genre  = (const float*)d_in[2];
    const float* W_hid  = (const float*)d_in[3];
    const float* b_hid  = (const float*)d_in[4];
    const float* emb    = (const float*)d_in[5];
    const float* W_ih1  = (const float*)d_in[6];
    const float* W_hh1  = (const float*)d_in[7];
    const float* b_ih1  = (const float*)d_in[8];
    const float* b_hh1  = (const float*)d_in[9];
    const float* W_ih2  = (const float*)d_in[10];
    const float* W_hh2  = (const float*)d_in[11];
    const float* b_ih2  = (const float*)d_in[12];
    const float* b_hh2  = (const float*)d_in[13];
    const float* W_out  = (const float*)d_in[14];
    const float* b_out  = (const float*)d_in[15];

    float* out   = (float*)d_out;                        // [B,T,NC] logits
    float* pred  = out + (size_t)BB * TT * NCC;          // [B,T] sampled tokens

    // workspace carve-out (~22 MB)
    char* p = (char*)d_ws;
    auto alloc = [&](size_t bytes) -> void* {
        void* r = (void*)p;
        p += (bytes + 255) & ~(size_t)255;
        return r;
    };
    unsigned short* emb_bf  = (unsigned short*)alloc((size_t)NCC * HH * 2);
    unsigned short* wih1_bf = (unsigned short*)alloc((size_t)H3 * HH * 2);
    unsigned short* whh1_bf = (unsigned short*)alloc((size_t)H3 * HH * 2);
    unsigned short* wih2_bf = (unsigned short*)alloc((size_t)H3 * HH * 2);
    unsigned short* whh2_bf = (unsigned short*)alloc((size_t)H3 * HH * 2);
    unsigned short* wout_bf = (unsigned short*)alloc((size_t)NCC * HH * 2);
    float*          h1_f    = (float*)alloc((size_t)BB * HH * 4);
    float*          h2_f    = (float*)alloc((size_t)BB * HH * 4);
    unsigned short* h1_bf   = (unsigned short*)alloc((size_t)BB * HH * 2);
    unsigned short* h2_bf   = (unsigned short*)alloc((size_t)BB * HH * 2);
    float*          gi1     = (float*)alloc((size_t)BB * H3 * 4);
    float*          gh1     = (float*)alloc((size_t)BB * H3 * 4);
    float*          gi2     = (float*)alloc((size_t)BB * H3 * 4);
    float*          gh2     = (float*)alloc((size_t)BB * H3 * 4);
    int*            tok     = (int*)alloc((size_t)BB * 4);

    // one-time: weight conversion (bf16 set ~7MB stays hot in 192MB L2)
    auto cvt = [&](const float* s, unsigned short* d, int n) {
        cvt_bf16_kernel<<<(n + 255) / 256, 256, 0, stream>>>(s, d, n);
    };
    cvt(emb,   emb_bf,  NCC * HH);
    cvt(W_ih1, wih1_bf, H3 * HH);
    cvt(W_hh1, whh1_bf, H3 * HH);
    cvt(W_ih2, wih2_bf, H3 * HH);
    cvt(W_hh2, whh2_bf, H3 * HH);
    cvt(W_out, wout_bf, NCC * HH);

    tok_init_kernel<<<(BB + 255) / 256, 256, 0, stream>>>(tok);
    hidden_init_kernel<<<(BB * HH) / 256, 256, 0, stream>>>(
        note, genre, W_hid, b_hid, h1_f, h1_bf);

    const dim3 gemm_block(32, 1, 1);
    const dim3 grid_dual(H3 / 64, BB / 16, 2);     // gi & gh in one launch
    const dim3 grid_log(NCC / 64, BB / 16, 1);     // N=512 logits GEMM
    const int  elems = BB * HH;

    for (int t = 0; t < TT; ++t) {
        // GRU1: z=0 -> gi1 = emb[tok] @ W_ih1^T + b_ih1 (gather fused)
        //       z=1 -> gh1 = h1 @ W_hh1^T + b_hh1
        wmma_gemm_dual_kernel<<<grid_dual, gemm_block, 0, stream>>>(
            emb_bf, tok, wih1_bf, b_ih1, gi1,
            h1_bf, whh1_bf, b_hh1, gh1, HH, H3);
        // gates -> h1 (at t==0 also h2 = h1, i.e. hx[1] = hx[0])
        gru_gates_kernel<<<elems / 256, 256, 0, stream>>>(
            gi1, gh1, h1_f, h1_bf,
            (t == 0) ? h2_f : nullptr, (t == 0) ? h2_bf : nullptr);

        // GRU2: z=0 -> gi2 = h1 @ W_ih2^T, z=1 -> gh2 = h2 @ W_hh2^T
        wmma_gemm_dual_kernel<<<grid_dual, gemm_block, 0, stream>>>(
            h1_bf, nullptr, wih2_bf, b_ih2, gi2,
            h2_bf, whh2_bf, b_hh2, gh2, HH, H3);
        gru_gates_kernel<<<elems / 256, 256, 0, stream>>>(
            gi2, gh2, h2_f, h2_bf, nullptr, nullptr);

        // logits = h2 @ W_out^T + b_out, written directly to out[:, t, :]
        wmma_gemm_kernel<<<grid_log, gemm_block, 0, stream>>>(
            h2_bf, nullptr, wout_bf, b_out, out + (size_t)t * NCC,
            HH, TT * NCC);

        // categorical sample -> next token
        sample_kernel<<<BB, 256, 0, stream>>>(out, t, tok, pred);
    }
}
